// GptOssAttention_19980187861073
// MI455X (gfx1250) — matrix operand order repaired and checked by reading, not compiled, not run on hardware
//
#include <hip/hip_runtime.h>
#include <stdint.h>

// ---------------------------------------------------------------------------
// Types for WMMA fragments (gfx1250, wave32)
// ---------------------------------------------------------------------------
typedef __attribute__((ext_vector_type(16))) __bf16         v16bf;
typedef __attribute__((ext_vector_type(8)))  float          v8f;
typedef __attribute__((ext_vector_type(8)))  unsigned short v8us;
typedef __attribute__((ext_vector_type(4)))  unsigned short v4us;
typedef __attribute__((ext_vector_type(4)))  int            v4i;

union B16Frag {            // 32 bytes = 8 VGPRs, one A or B bf16 fragment
    v16bf         v;
    v8us          h[2];    // h[0] = elements 0..7, h[1] = elements 8..15
    unsigned short s[16];
};

__device__ __forceinline__ unsigned short f32_to_bf16(float f) {
    union { float f; unsigned int u; } x; x.f = f;
    unsigned int r = x.u + 0x7FFFu + ((x.u >> 16) & 1u);   // round-to-nearest-even
    return (unsigned short)(r >> 16);
}

// ---------------------------------------------------------------------------
// Async global->LDS copy (CDNA5 GLOBAL_LOAD_ASYNC_TO_LDS_B128, ASYNCcnt).
// Builtin signature (from hipcc diagnostic): (global int4*, local int4*, imm, imm)
// Guarded: falls back to a VGPR round-trip if the builtin is unavailable.
// ---------------------------------------------------------------------------
#if __has_builtin(__builtin_amdgcn_global_load_async_to_lds_b128)
#define HAS_ASYNC_COPY 1
__device__ __forceinline__ void cp16_to_lds(void* ldst, const void* gsrc) {
    __builtin_amdgcn_global_load_async_to_lds_b128(
        (__attribute__((address_space(1))) v4i*)(v4i*)(void*)gsrc,
        (__attribute__((address_space(3))) v4i*)(v4i*)ldst,
        0, 0);
}
__device__ __forceinline__ void wait_async_copies() {
#if __has_builtin(__builtin_amdgcn_s_wait_asynccnt)
    __builtin_amdgcn_s_wait_asynccnt(0);
#else
    asm volatile("s_wait_asynccnt 0x0" ::: "memory");
#endif
}
#else
#define HAS_ASYNC_COPY 0
__device__ __forceinline__ void cp16_to_lds(void* ldst, const void* gsrc) {
    *(v8us*)ldst = *(const v8us*)gsrc;
}
__device__ __forceinline__ void wait_async_copies() {}
#endif

// ---------------------------------------------------------------------------
// Problem dimensions (fixed by the reference)
// ---------------------------------------------------------------------------
#define S_LEN 2048
#define HID   2880
#define NH    64
#define NKV   8
#define HD    64
#define QDIM  (NH * HD)    // 4096
#define KVDIM (NKV * HD)   // 512

// ---------------------------------------------------------------------------
// f32 -> bf16 cast (vectorized by 4)
// ---------------------------------------------------------------------------
__global__ void cast_f32_to_bf16(const float* __restrict__ in,
                                 unsigned short* __restrict__ out, int n4) {
    int i = blockIdx.x * blockDim.x + threadIdx.x;
    if (i >= n4) return;
    float4 f = reinterpret_cast<const float4*>(in)[i];
    v4us o;
    o[0] = f32_to_bf16(f.x);
    o[1] = f32_to_bf16(f.y);
    o[2] = f32_to_bf16(f.z);
    o[3] = f32_to_bf16(f.w);
    reinterpret_cast<v4us*>(out)[i] = o;
}

// ---------------------------------------------------------------------------
// Generic bf16 WMMA GEMM:  C[M,N] = A[M,K] * W[N,K]^T + bias[N]
// Block tile 128(M) x 64(N), 8 waves in a 4x2 grid of 32x32 wave tiles.
// Double-buffered LDS slabs filled with async global->LDS copies.
// OUT_MODE: 0 = bf16 [M,N], 1 = bf16 transposed [N,M], 2 = f32 [M,N]
// ---------------------------------------------------------------------------
template <int OUT_MODE>
__global__ __launch_bounds__(256)
void gemm_bf16_wmma(const unsigned short* __restrict__ A,
                    const unsigned short* __restrict__ W,
                    const float* __restrict__ bias,
                    unsigned short* __restrict__ outB,
                    float* __restrict__ outF,
                    int M, int N, int K) {
    __shared__ unsigned short At[2][128 * 40];   // rows padded to 40 (80 B, 16B-aligned)
    __shared__ unsigned short Bt[2][64 * 40];    // stored as [n][k]

    const int t    = threadIdx.x;
    const int lane = t & 31;
    const int w    = t >> 5;
    const int wm   = w & 3;          // 4 wave rows   (32 M each)
    const int wn   = w >> 2;         // 2 wave cols   (32 N each)
    const int nl   = lane & 15;
    const int half = lane >> 4;

    const int m0 = blockIdx.y * 128;
    const int n0 = blockIdx.x * 64;

    v8f c[2][2] = {};

    // cooperative tile-load mapping
    const int arow = t >> 1, achunk = t & 1;       // 128 rows x 2 chunks of 16
    const int brow = t >> 2, bchunk = t & 3;       // 64 rows  x 4 chunks of 8
    const unsigned short* aSrc = A + (size_t)(m0 + arow) * K + achunk * 16;
    const unsigned short* bSrc = W + (size_t)(n0 + brow) * K + bchunk * 8;
    const int aOff = arow * 40 + achunk * 16;
    const int bOff = brow * 40 + bchunk * 8;

    // prologue: issue slab 0 into buffer 0
    cp16_to_lds(&At[0][aOff],     aSrc);
    cp16_to_lds(&At[0][aOff + 8], aSrc + 8);
    cp16_to_lds(&Bt[0][bOff],     bSrc);

    int ib = 0;
    #pragma unroll 1
    for (int k0 = 0; k0 < K; k0 += 32, ib ^= 1) {
        wait_async_copies();       // this wave's slab-ib copies have landed
        __syncthreads();           // all waves' copies done; prior reads retired

        if (k0 + 32 < K) {         // issue next slab into the other buffer
            cp16_to_lds(&At[ib ^ 1][aOff],     aSrc + k0 + 32);
            cp16_to_lds(&At[ib ^ 1][aOff + 8], aSrc + k0 + 40);
            cp16_to_lds(&Bt[ib ^ 1][bOff],     bSrc + k0 + 32);
        }

        B16Frag a[2], b[2];
        #pragma unroll
        for (int i = 0; i < 2; ++i) {
            const unsigned short* p = &At[ib][(wm * 32 + i * 16 + nl) * 40 + half * 8];
            a[i].h[0] = *(const v8us*)(p);        // K = half*8 .. +7
            a[i].h[1] = *(const v8us*)(p + 16);   // K = half*8+16 .. +7
        }
        #pragma unroll
        for (int j = 0; j < 2; ++j) {
            const unsigned short* p = &Bt[ib][(wn * 32 + j * 16 + nl) * 40 + half * 8];
            b[j].h[0] = *(const v8us*)(p);        // K = half*8 .. +7
            b[j].h[1] = *(const v8us*)(p + 16);   // K = half*8+16 .. +7
        }
        #pragma unroll
        for (int i = 0; i < 2; ++i)
            #pragma unroll
            for (int j = 0; j < 2; ++j)
                c[i][j] = __builtin_amdgcn_wmma_f32_16x16x32_bf16(
                    false, a[i].v, false, b[j].v, (short)0, c[i][j], false, false);
    }

    // epilogue: + bias, store (compile-time output mode)
    #pragma unroll
    for (int j = 0; j < 2; ++j) {
        const int ng = n0 + wn * 32 + j * 16 + nl;
        const float bv = bias[ng];
        #pragma unroll
        for (int i = 0; i < 2; ++i) {
            #pragma unroll
            for (int r = 0; r < 8; ++r) {
                const int mg = m0 + wm * 32 + i * 16 + r + 8 * half;
                const float v = c[i][j][r] + bv;
                if (OUT_MODE == 2)      outF[(size_t)mg * N + ng] = v;
                else if (OUT_MODE == 1) outB[(size_t)ng * M + mg] = f32_to_bf16(v);
                else                    outB[(size_t)mg * N + ng] = f32_to_bf16(v);
            }
        }
    }
}

// ---------------------------------------------------------------------------
// Flash attention, one wave = one (head, 16-row q strip); 8 waves / block.
// K stored [s][kv*64+d] (row-major), V stored transposed Vt[kv*64+d][s].
// Causal mask applied analytically (== tril/-1e9 additive mask).
// ---------------------------------------------------------------------------
__global__ __launch_bounds__(256)
void attn_fwd_wmma(const unsigned short* __restrict__ Q,
                   const unsigned short* __restrict__ Kmat,
                   const unsigned short* __restrict__ Vt,
                   unsigned short* __restrict__ Obuf) {
    __shared__ unsigned short ldsP[8 * 16 * 32];   // per-wave P staging

    const int lane = threadIdx.x & 31;
    const int w    = threadIdx.x >> 5;
    const int nl   = lane & 15;
    const int half = lane >> 4;
    const int h    = blockIdx.y;
    const int kvh  = h >> 3;                 // G = 8
    const int q0   = blockIdx.x * 128 + w * 16;

    // Q strip as two A-fragments (d = 0..31 and 32..63)
    B16Frag qa[2];
    #pragma unroll
    for (int db = 0; db < 2; ++db) {
        const unsigned short* p =
            Q + (size_t)(q0 + nl) * QDIM + h * HD + db * 32 + half * 8;
        qa[db].h[0] = *(const v8us*)(p);
        qa[db].h[1] = *(const v8us*)(p + 16);
    }

    v8f o[4] = {};
    float mst[8], lst[8];
    #pragma unroll
    for (int r = 0; r < 8; ++r) { mst[r] = -3.0e38f; lst[r] = 0.0f; }

    unsigned short* myP = ldsP + w * 512;
    const int nkt = (q0 + 16 + 31) >> 5;     // 32-key tiles covering causal span

    #pragma unroll 1
    for (int kt = 0; kt < nkt; ++kt) {
        const int key0 = kt * 32;

        // ---- scores: S[16 x 32] = Q(16x64) . K^T ----
        v8f c[2] = {};
        #pragma unroll
        for (int tcol = 0; tcol < 2; ++tcol) {
            #pragma unroll
            for (int db = 0; db < 2; ++db) {
                B16Frag bk;
                const unsigned short* p = Kmat +
                    (size_t)(key0 + tcol * 16 + nl) * KVDIM + kvh * HD + db * 32 + half * 8;
                bk.h[0] = *(const v8us*)(p);
                bk.h[1] = *(const v8us*)(p + 16);
                c[tcol] = __builtin_amdgcn_wmma_f32_16x16x32_bf16(
                    false, qa[db].v, false, bk.v, (short)0, c[tcol], false, false);
            }
        }

        // ---- scale + causal mask + online softmax ----
        float p0[8], p1[8], alpha[8];
        #pragma unroll
        for (int r = 0; r < 8; ++r) {
            const int mrow = q0 + r + 8 * half;
            const int k0c  = key0 + nl;
            const int k1c  = key0 + 16 + nl;
            float s0 = c[0][r] * 0.125f + (k0c > mrow ? -1.0e9f : 0.0f);
            float s1 = c[1][r] * 0.125f + (k1c > mrow ? -1.0e9f : 0.0f);
            float mx = fmaxf(s0, s1);
            mx = fmaxf(mx, __shfl_xor(mx, 1, 32));
            mx = fmaxf(mx, __shfl_xor(mx, 2, 32));
            mx = fmaxf(mx, __shfl_xor(mx, 4, 32));
            mx = fmaxf(mx, __shfl_xor(mx, 8, 32));
            const float mnew = fmaxf(mst[r], mx);
            const float a    = __expf(mst[r] - mnew);
            const float e0   = __expf(s0 - mnew);
            const float e1   = __expf(s1 - mnew);
            float sum = e0 + e1;
            sum += __shfl_xor(sum, 1, 32);
            sum += __shfl_xor(sum, 2, 32);
            sum += __shfl_xor(sum, 4, 32);
            sum += __shfl_xor(sum, 8, 32);
            lst[r]   = lst[r] * a + sum;
            mst[r]   = mnew;
            alpha[r] = a;
            p0[r] = e0;
            p1[r] = e1;
        }

        // rescale O accumulators
        #pragma unroll
        for (int f = 0; f < 4; ++f)
            #pragma unroll
            for (int r = 0; r < 8; ++r)
                o[f][r] *= alpha[r];

        // ---- re-layout P (C-frag -> A-frag) through wave-private LDS ----
        #pragma unroll
        for (int r = 0; r < 8; ++r) {
            const int m = r + 8 * half;
            myP[m * 32 + nl]      = f32_to_bf16(p0[r]);
            myP[m * 32 + 16 + nl] = f32_to_bf16(p1[r]);
        }
        B16Frag pa;
        {
            const unsigned short* pp = myP + nl * 32 + half * 8;
            pa.h[0] = *(const v8us*)(pp);
            pa.h[1] = *(const v8us*)(pp + 16);
        }

        // ---- O += P(16x32) . V(32x64), 4 column chunks of 16 ----
        #pragma unroll
        for (int cd = 0; cd < 4; ++cd) {
            B16Frag bv;
            const unsigned short* p = Vt +
                (size_t)(kvh * HD + cd * 16 + nl) * S_LEN + key0 + half * 8;
            bv.h[0] = *(const v8us*)(p);
            bv.h[1] = *(const v8us*)(p + 16);
            o[cd] = __builtin_amdgcn_wmma_f32_16x16x32_bf16(
                false, pa.v, false, bv.v, (short)0, o[cd], false, false);
        }
    }

    // ---- finalize: O / l, store bf16 into [s, h*64+d] ----
    #pragma unroll
    for (int cd = 0; cd < 4; ++cd) {
        #pragma unroll
        for (int r = 0; r < 8; ++r) {
            const int row = q0 + r + 8 * half;
            const float v = o[cd][r] / lst[r];
            Obuf[(size_t)row * QDIM + h * HD + cd * 16 + nl] = f32_to_bf16(v);
        }
    }
}

// ---------------------------------------------------------------------------
// Host-side orchestration
// ---------------------------------------------------------------------------
extern "C" void kernel_launch(void* const* d_in, const int* in_sizes, int n_in,
                              void* d_out, int out_size, void* d_ws, size_t ws_size,
                              hipStream_t stream) {
    (void)in_sizes; (void)n_in; (void)out_size; (void)ws_size;

    const float* hs = (const float*)d_in[0];
    // d_in[1] = attention_mask (pure causal -1e9; applied analytically)
    const float* Wq = (const float*)d_in[2];
    const float* bq = (const float*)d_in[3];
    const float* Wk = (const float*)d_in[4];
    const float* bk = (const float*)d_in[5];
    const float* Wv = (const float*)d_in[6];
    const float* bv = (const float*)d_in[7];
    const float* Wo = (const float*)d_in[8];
    const float* bo = (const float*)d_in[9];
    float* out = (float*)d_out;

    // workspace layout (bf16 = unsigned short)
    char* ws = (char*)d_ws;
    size_t off = 0;
    unsigned short* hsB = (unsigned short*)(ws + off); off += (size_t)S_LEN * HID * 2;    // 11.8 MB
    unsigned short* WqB = (unsigned short*)(ws + off); off += (size_t)QDIM * HID * 2;     // 23.6 MB
    unsigned short* WkB = (unsigned short*)(ws + off); off += (size_t)KVDIM * HID * 2;    //  3.0 MB
    unsigned short* WvB = (unsigned short*)(ws + off); off += (size_t)KVDIM * HID * 2;    //  3.0 MB
    unsigned short* WoB = (unsigned short*)(ws + off); off += (size_t)HID * QDIM * 2;     // 23.6 MB
    unsigned short* Qb  = (unsigned short*)(ws + off); off += (size_t)S_LEN * QDIM * 2;   // 16.8 MB
    unsigned short* Kb  = (unsigned short*)(ws + off); off += (size_t)S_LEN * KVDIM * 2;  //  2.1 MB
    unsigned short* VtB = (unsigned short*)(ws + off); off += (size_t)KVDIM * S_LEN * 2;  //  2.1 MB
    unsigned short* Ab  = (unsigned short*)(ws + off); off += (size_t)S_LEN * QDIM * 2;   // 16.8 MB

    // 1) downconvert inputs/weights to bf16
    auto cast = [&](const float* src, unsigned short* dst, size_t n) {
        int n4 = (int)(n / 4);
        cast_f32_to_bf16<<<(n4 + 255) / 256, 256, 0, stream>>>(src, dst, n4);
    };
    cast(hs, hsB, (size_t)S_LEN * HID);
    cast(Wq, WqB, (size_t)QDIM * HID);
    cast(Wk, WkB, (size_t)KVDIM * HID);
    cast(Wv, WvB, (size_t)KVDIM * HID);
    cast(Wo, WoB, (size_t)HID * QDIM);

    // 2) Q / K / V projections (V stored transposed [d][s] for attention)
    gemm_bf16_wmma<0><<<dim3(QDIM / 64, S_LEN / 128), 256, 0, stream>>>(
        hsB, WqB, bq, Qb, nullptr, S_LEN, QDIM, HID);
    gemm_bf16_wmma<0><<<dim3(KVDIM / 64, S_LEN / 128), 256, 0, stream>>>(
        hsB, WkB, bk, Kb, nullptr, S_LEN, KVDIM, HID);
    gemm_bf16_wmma<1><<<dim3(KVDIM / 64, S_LEN / 128), 256, 0, stream>>>(
        hsB, WvB, bv, VtB, nullptr, S_LEN, KVDIM, HID);

    // 3) flash attention: grid = (q blocks of 128, heads)
    attn_fwd_wmma<<<dim3(S_LEN / 128, NH), 256, 0, stream>>>(Qb, Kb, VtB, Ab);

    // 4) output projection -> f32 d_out
    gemm_bf16_wmma<2><<<dim3(HID / 64, S_LEN / 128), 256, 0, stream>>>(
        Ab, WoB, bo, nullptr, out, S_LEN, HID, QDIM);
}
